// AttentionHead_36799279792242
// MI455X (gfx1250) — matrix operand order repaired and checked by reading, not compiled, not run on hardware
//
#include <hip/hip_runtime.h>
#include <hip/hip_bf16.h>

typedef __attribute__((ext_vector_type(16))) __bf16 v16bf;
typedef __attribute__((ext_vector_type(8)))  __bf16 v8bf;
typedef __attribute__((ext_vector_type(4)))  __bf16 v4bf;
typedef __attribute__((ext_vector_type(8)))  float  v8f;
typedef __attribute__((ext_vector_type(4)))  unsigned int u32x4;
typedef __attribute__((ext_vector_type(8)))  int i32x8;
typedef __attribute__((ext_vector_type(4)))  int i32x4;

#define HEAD_DIM 128
#define SEQ      2048
#define NBATCH   8
#define TILE_K   64
#define NTILES   (SEQ / TILE_K)
#define WAVES    8          // 8 waves x 16 q-rows -> 128 q rows per block
#define KSTRIDE  144        // TDM pad: 128 elems + 8 dwords pad -> 144 (288B rows)
#define VSTRIDE  80         // TDM pad: 64 elems + 8 dwords pad -> 80 (160B rows)
#define PSTRIDE  72
#define NEGMASK  (-1000000000.0f)

// ---------------- Prep: K -> bf16 row-major, V -> bf16 transposed [b][d][s] ----------------
__global__ __launch_bounds__(256)
void cvt_kernel(const float* __restrict__ K, const float* __restrict__ V,
                __bf16* __restrict__ Kb, __bf16* __restrict__ Vtb) {
    const int idx = blockIdx.x * 256 + threadIdx.x;     // one float4 chunk
    const int b   = idx >> 16;                          // 2048*32 chunks per batch
    const int rem = idx & 65535;
    const int s   = rem >> 5;
    const int c4  = (rem & 31) << 2;
    const size_t g = ((size_t)b * SEQ + (size_t)s) * HEAD_DIM + c4;
    const float4 kq = *(const float4*)(K + g);
    const float4 vq = *(const float4*)(V + g);
    v4bf kp;
    kp[0]=(__bf16)kq.x; kp[1]=(__bf16)kq.y; kp[2]=(__bf16)kq.z; kp[3]=(__bf16)kq.w;
    *(v4bf*)(Kb + g) = kp;
    const size_t vb = (size_t)b * HEAD_DIM * SEQ;
    Vtb[vb + (size_t)(c4 + 0) * SEQ + s] = (__bf16)vq.x;
    Vtb[vb + (size_t)(c4 + 1) * SEQ + s] = (__bf16)vq.y;
    Vtb[vb + (size_t)(c4 + 2) * SEQ + s] = (__bf16)vq.z;
    Vtb[vb + (size_t)(c4 + 3) * SEQ + s] = (__bf16)vq.w;
}

// ---------------- TDM descriptor helpers (D# per CDNA5 ISA ch.8) ----------------
__device__ __forceinline__ u32x4 tdm_group0(unsigned lds_addr, unsigned long long gva) {
    u32x4 g;
    g[0] = 1u;                                    // count=1 valid descriptor
    g[1] = lds_addr;                              // lds_addr[31:0]
    g[2] = (unsigned)(gva & 0xFFFFFFFFull);       // global_addr[31:0]
    g[3] = ((unsigned)(gva >> 32) & 0x01FFFFFFu)  // global_addr[56:32]
         | (2u << 30);                            // type = 2 (image)
    return g;
}
// 2D tensor group1: data_size=2B, pad_enable, tile (t0 x t1), stride0, dims
__device__ __forceinline__ i32x8 tdm_group1(unsigned pad_interval, unsigned pad_amount,
                                            unsigned tensor_d0, unsigned tensor_d1,
                                            unsigned tile_d0, unsigned tile_d1,
                                            unsigned stride0) {
    i32x8 g;
    g[0] = (int)((1u << 16) | (1u << 20) | (pad_interval << 22) | (pad_amount << 25));
    g[1] = (int)((tensor_d0 & 0xFFFFu) << 16);              // dim0[15:0] -> bits 63:48
    g[2] = (int)(((tensor_d0 >> 16) & 0xFFFFu) | ((tensor_d1 & 0xFFFFu) << 16));
    g[3] = (int)(((tensor_d1 >> 16) & 0xFFFFu) | ((tile_d0 & 0xFFFFu) << 16));
    g[4] = (int)(tile_d1 & 0xFFFFu);                        // tile_dim1; tile_dim2=0
    g[5] = (int)stride0;                                    // dim0_stride[31:0]
    g[6] = 0;                                               // stride0 hi / stride1 lo
    g[7] = 0;
    return g;
}

// ---------------- Main flash-attention kernel ----------------
__global__ __launch_bounds__(256)
void attn_fa_wmma_kernel(const float* __restrict__ Q,
                         const __bf16* __restrict__ Kb,
                         const __bf16* __restrict__ Vtb,
                         const int*   __restrict__ lens,
                         float*       __restrict__ O) {
    __shared__ alignas(32) __bf16 sKb[2][TILE_K * KSTRIDE];    // [kv][d], TDM-padded
    __shared__ alignas(32) __bf16 sVb[2][HEAD_DIM * VSTRIDE];  // [d][kv], TDM-padded
    __shared__ alignas(32) __bf16 sP[WAVES][16 * PSTRIDE];     // per-wave [q][kv]
    __shared__ alignas(32) float  sBc[WAVES][16];

    const int b     = blockIdx.y;
    const int t     = threadIdx.x;
    const int wave  = t >> 5;
    const int lane  = t & 31;
    const int lhalf = lane >> 4;
    const int ln    = lane & 15;
    const int q0    = blockIdx.x * (16 * WAVES) + wave * 16;
    const int len   = lens[b];
    const float scale2 = 0.088388347648318447f * 1.4426950408889634f; // /sqrt(d) * log2(e)

    // Batch-local tensor bases + LDS buffer addresses (uniform)
    const unsigned long long kva0 =
        (unsigned long long)(uintptr_t)(Kb + (size_t)b * SEQ * HEAD_DIM);
    const unsigned long long vva0 =
        (unsigned long long)(uintptr_t)(Vtb + (size_t)b * HEAD_DIM * SEQ);
    const unsigned ldsK0 = (unsigned)(uintptr_t)&sKb[0][0];
    const unsigned ldsK1 = (unsigned)(uintptr_t)&sKb[1][0];
    const unsigned ldsV0 = (unsigned)(uintptr_t)&sVb[0][0];
    const unsigned ldsV1 = (unsigned)(uintptr_t)&sVb[1][0];
    // K tile: rows=64 kv, row=128 elems (64 dwords) -> pad_interval=5, pad 8 dwords
    const i32x8 g1K = tdm_group1(5u, 7u, HEAD_DIM, SEQ, HEAD_DIM, TILE_K, HEAD_DIM);
    // V tile: rows=128 d, row=64 elems (32 dwords) -> pad_interval=4, pad 8 dwords
    const i32x8 g1V = tdm_group1(4u, 7u, SEQ, HEAD_DIM, TILE_K, HEAD_DIM, SEQ);
    const i32x4 zg  = {};
    const i32x8 zg8 = {};

    // ---- Prefetch tile 0 into buffer 0 (wave 0 only; TDM ignores EXEC) ----
    if (t < 32) {
        __builtin_amdgcn_tensor_load_to_lds(tdm_group0(ldsK0, kva0), g1K, zg, zg, zg8, 0);
        __builtin_amdgcn_tensor_load_to_lds(tdm_group0(ldsV0, vva0), g1V, zg, zg, zg8, 0);
    }

    // ---- Persistent Q^T B-fragments (from fp32 global, once) ----
    v16bf qf[4];
    {
        const float* qrow = Q + ((size_t)b * SEQ + (size_t)(q0 + ln)) * HEAD_DIM;
        #pragma unroll
        for (int c = 0; c < 4; ++c) {
            const int dbase = c * 32 + lhalf * 16;
            const float4 f0 = *(const float4*)(qrow + dbase + 0);
            const float4 f1 = *(const float4*)(qrow + dbase + 4);
            const float4 f2 = *(const float4*)(qrow + dbase + 8);
            const float4 f3 = *(const float4*)(qrow + dbase + 12);
            v16bf a;
            a[0]=(__bf16)f0.x;  a[1]=(__bf16)f0.y;  a[2]=(__bf16)f0.z;  a[3]=(__bf16)f0.w;
            a[4]=(__bf16)f1.x;  a[5]=(__bf16)f1.y;  a[6]=(__bf16)f1.z;  a[7]=(__bf16)f1.w;
            a[8]=(__bf16)f2.x;  a[9]=(__bf16)f2.y;  a[10]=(__bf16)f2.z; a[11]=(__bf16)f2.w;
            a[12]=(__bf16)f3.x; a[13]=(__bf16)f3.y; a[14]=(__bf16)f3.z; a[15]=(__bf16)f3.w;
            qf[c] = a;
        }
    }

    const bool qok = (q0 + ln) < len;
    float m_s = -INFINITY;
    float l_s = 0.0f;
    v8f o_acc[8];
    #pragma unroll
    for (int d = 0; d < 8; ++d) { v8f z = {}; o_acc[d] = z; }

    for (int kt = 0; kt < NTILES; ++kt) {
        const int k0  = kt * TILE_K;
        const int cur = kt & 1;
        __syncthreads();   // all waves done with buffer cur^1 (iter kt-1)

        if (t < 32) {
            if (kt + 1 < NTILES) {  // prefetch next tile into the other buffer
                const unsigned long long kva = kva0 +
                    (unsigned long long)(kt + 1) * TILE_K * HEAD_DIM * sizeof(__bf16);
                const unsigned long long vva = vva0 +
                    (unsigned long long)(kt + 1) * TILE_K * sizeof(__bf16);
                const unsigned lk = (cur ? ldsK0 : ldsK1);
                const unsigned lv = (cur ? ldsV0 : ldsV1);
                __builtin_amdgcn_tensor_load_to_lds(tdm_group0(lk, kva), g1K, zg, zg, zg8, 0);
                __builtin_amdgcn_tensor_load_to_lds(tdm_group0(lv, vva), g1V, zg, zg, zg8, 0);
                __builtin_amdgcn_s_wait_tensorcnt(2);  // in-order: tile kt has landed
            } else {
                __builtin_amdgcn_s_wait_tensorcnt(0);
            }
        }
        __syncthreads();   // tile kt visible to all waves

        // ---- S^T = K Q^T ----
        v8f sc[4];
        #pragma unroll
        for (int nt = 0; nt < 4; ++nt) {
            v8f acc = {};
            const __bf16* krow = &sKb[cur][(nt * 16 + ln) * KSTRIDE];
            #pragma unroll
            for (int c = 0; c < 4; ++c) {
                const v8bf lo = *(const v8bf*)(krow + c * 32 + lhalf * 16);
                const v8bf hi = *(const v8bf*)(krow + c * 32 + lhalf * 16 + 8);
                v16bf ka;
                #pragma unroll
                for (int e = 0; e < 8; ++e) { ka[e] = lo[e]; ka[8 + e] = hi[e]; }
                acc = __builtin_amdgcn_wmma_f32_16x16x32_bf16(
                        false, ka, false, qf[c], (short)0, acc, false, false);
            }
            #pragma unroll
            for (int v = 0; v < 8; ++v) {
                const int kv = k0 + nt * 16 + lhalf * 8 + v;
                float s = acc[v] * scale2;
                if (!qok || kv >= len) s = NEGMASK;
                acc[v] = s;
            }
            sc[nt] = acc;
        }

        // ---- Column softmax: in-lane tree + one cross-half exchange ----
        v8f tmax;
        #pragma unroll
        for (int v = 0; v < 8; ++v)
            tmax[v] = fmaxf(fmaxf(sc[0][v], sc[1][v]), fmaxf(sc[2][v], sc[3][v]));
        float cm = tmax[0];
        #pragma unroll
        for (int v = 1; v < 8; ++v) cm = fmaxf(cm, tmax[v]);
        cm = fmaxf(cm, __shfl_xor(cm, 16, 32));

        const float mnew = fmaxf(m_s, cm);
        const float fac  = __builtin_amdgcn_exp2f(m_s - mnew);
        #pragma unroll
        for (int nt = 0; nt < 4; ++nt)
            #pragma unroll
            for (int v = 0; v < 8; ++v)
                sc[nt][v] = __builtin_amdgcn_exp2f(sc[nt][v] - mnew);
        v8f tsum;
        #pragma unroll
        for (int v = 0; v < 8; ++v)
            tsum[v] = (sc[0][v] + sc[1][v]) + (sc[2][v] + sc[3][v]);
        float rs = tsum[0];
        #pragma unroll
        for (int v = 1; v < 8; ++v) rs += tsum[v];
        rs += __shfl_xor(rs, 16, 32);
        l_s = l_s * fac + rs;
        m_s = mnew;

        // ---- Spill P^T into [q][kv] LDS (packed b128) + fac broadcast ----
        if (lhalf == 0) sBc[wave][ln] = fac;
        #pragma unroll
        for (int nt = 0; nt < 4; ++nt) {
            v8bf pb;
            #pragma unroll
            for (int v = 0; v < 8; ++v) pb[v] = (__bf16)sc[nt][v];
            *(v8bf*)(&sP[wave][ln * PSTRIDE + nt * 16 + lhalf * 8]) = pb;
        }
        __syncthreads();

        // ---- O = diag(fac)*O + P V ----
        const float4 fb0 = *(const float4*)(&sBc[wave][lhalf * 8]);
        const float4 fb1 = *(const float4*)(&sBc[wave][lhalf * 8 + 4]);
        const float facv[8] = { fb0.x, fb0.y, fb0.z, fb0.w, fb1.x, fb1.y, fb1.z, fb1.w };

        v16bf pa[2];
        {
            const __bf16* pr = sP[wave] + ln * PSTRIDE;
            #pragma unroll
            for (int kc = 0; kc < 2; ++kc) {
                const v8bf lo = *(const v8bf*)(pr + kc * 32 + lhalf * 16);
                const v8bf hi = *(const v8bf*)(pr + kc * 32 + lhalf * 16 + 8);
                v16bf a;
                #pragma unroll
                for (int e = 0; e < 8; ++e) { a[e] = lo[e]; a[8 + e] = hi[e]; }
                pa[kc] = a;
            }
        }
        #pragma unroll
        for (int dt = 0; dt < 8; ++dt) {
            v8f c = o_acc[dt];
            #pragma unroll
            for (int v = 0; v < 8; ++v) c[v] *= facv[v];
            #pragma unroll
            for (int kc = 0; kc < 2; ++kc) {
                const v16bf bv = *(const v16bf*)
                    (&sVb[cur][(dt * 16 + ln) * VSTRIDE + kc * 32 + lhalf * 16]);
                c = __builtin_amdgcn_wmma_f32_16x16x32_bf16(
                        false, pa[kc], false, bv, (short)0, c, false, false);
            }
            o_acc[dt] = c;
        }
    }

    // ---- Finalize ----
    __syncthreads();
    if (lhalf == 0) sBc[wave][ln] = 1.0f / l_s;
    __syncthreads();
    const float4 r0 = *(const float4*)(&sBc[wave][lhalf * 8]);
    const float4 r1 = *(const float4*)(&sBc[wave][lhalf * 8 + 4]);
    const float rinv[8] = { r0.x, r0.y, r0.z, r0.w, r1.x, r1.y, r1.z, r1.w };
    #pragma unroll
    for (int dt = 0; dt < 8; ++dt) {
        #pragma unroll
        for (int v = 0; v < 8; ++v) {
            const int qr = q0 + lhalf * 8 + v;
            O[((size_t)b * SEQ + (size_t)qr) * HEAD_DIM + dt * 16 + ln] =
                o_acc[dt][v] * rinv[v];
        }
    }
}

extern "C" void kernel_launch(void* const* d_in, const int* in_sizes, int n_in,
                              void* d_out, int out_size, void* d_ws, size_t ws_size,
                              hipStream_t stream) {
    const float* q    = (const float*)d_in[0];
    const float* k    = (const float*)d_in[1];
    const float* v    = (const float*)d_in[2];
    const int*   lens = (const int*)d_in[3];
    float*       out  = (float*)d_out;

    __bf16* kb  = (__bf16*)d_ws;                                    // 4 MB
    __bf16* vtb = kb + (size_t)NBATCH * SEQ * HEAD_DIM;             // 4 MB

    cvt_kernel<<<dim3(NBATCH * SEQ * HEAD_DIM / 4 / 256), dim3(256), 0, stream>>>(
        k, v, kb, vtb);

    dim3 grid(SEQ / (16 * WAVES), NBATCH);   // (16, 8)
    dim3 block(32 * WAVES);                  // 256 threads = 8 waves
    attn_fa_wmma_kernel<<<grid, block, 0, stream>>>(q, kb, vtb, lens, out);
}
